// TGL_24369644437996
// MI455X (gfx1250) — compile-verified, matrix operand order checked by reading, not run on hardware
//
#include <hip/hip_runtime.h>
#include <hip/hip_bf16.h>
#include <math.h>

// ---------------------------------------------------------------------------
// GCN (4 layers) + fusion + graph pooling for MI455X (gfx1250, wave32).
// Dense linears run on v_wmma_f32_16x16x32_bf16 with weights pre-swizzled
// into per-lane fragment order (vector b128 loads, no LDS scalar traffic).
// Edge scatter (HBM/L2-bound part) is float4 streaming + global_atomic_add_f32;
// its ~77MB working set fits the 192MB L2.
// ---------------------------------------------------------------------------

typedef __bf16 bf16;
typedef __attribute__((ext_vector_type(16))) __bf16 v16bf;
typedef __attribute__((ext_vector_type(8)))  float  v8f;

#define DHID 64

__device__ __forceinline__ float fast_tanh(float x) {
#if __has_builtin(__builtin_amdgcn_tanhf)
  return __builtin_amdgcn_tanhf(x);      // v_tanh_f32 on gfx1250
#else
  return tanhf(x);
#endif
}

// ---- monotonic float<->uint mapping for atomic segment-max ----------------
__device__ __forceinline__ unsigned f2ord(float f) {
  unsigned u = __float_as_uint(f);
  return (u & 0x80000000u) ? ~u : (u | 0x80000000u);
}
__device__ __forceinline__ float ord2f(unsigned u) {
  return (u & 0x80000000u) ? __uint_as_float(u & 0x7fffffffu)
                           : __uint_as_float(~u);
}

// ---- small utility kernels -------------------------------------------------
__global__ void fill_u32(unsigned* __restrict__ p, unsigned v, size_t n) {
  size_t i = (size_t)blockIdx.x * blockDim.x + threadIdx.x;
  if (i < n) p[i] = v;
}

__global__ void deg_init(float* __restrict__ deg, int n) {
  int i = blockIdx.x * blockDim.x + threadIdx.x;
  if (i < n) deg[i] = 1.0f;                       // self-loop contribution
}

__global__ void deg_accum(const int* __restrict__ col, float* __restrict__ deg, int e) {
  int i = blockIdx.x * blockDim.x + threadIdx.x;
  if (i < e) atomicAdd(&deg[col[i]], 1.0f);
}

__global__ void deg_rsqrt(float* __restrict__ deg, int n) {
  int i = blockIdx.x * blockDim.x + threadIdx.x;
  if (i < n) deg[i] = rsqrtf(deg[i]);             // in-place: deg -> dis
}

// ---- weight pre-swizzle into WMMA B-fragment order -------------------------
// Fragment ft = s*4 + tile covers K = 32s..32s+31, N = 16*tile..16*tile+15.
// Within a fragment: lane l, element e  ->  K = 32s + 8*(l>>4) + (e<8?e:e+8),
//                                           N = (l&15) + 16*tile.
// Packed layout: out[((ft*32) + l)*16 + e]  (32B per lane -> b128 loads).
template <int KIN>
__global__ void pack_weights(const float* __restrict__ W, bf16* __restrict__ out) {
  constexpr int KSTEPS = (KIN + 31) / 32;
  constexpr int TOTAL  = KSTEPS * 4 * 32 * 16;
  int t = blockIdx.x * blockDim.x + threadIdx.x;
  if (t >= TOTAL) return;
  int e    = t & 15;
  int l    = (t >> 4) & 31;
  int ft   = t >> 9;
  int tile = ft & 3;
  int s    = ft >> 2;
  int k = 32 * s + ((l >> 4) ? 8 : 0) + ((e < 8) ? e : (e + 8));
  int n = (l & 15) + 16 * tile;
  out[t] = (k < KIN) ? (bf16)W[k * DHID + n] : (bf16)0.0f;
}

// ---- node-wise linear via WMMA --------------------------------------------
// out[N,64] = act(in[N,KIN]) @ W[KIN,64] + b ; act = optional tanh and
// optional elementwise multiply by mul[N,64] (fusion layer).
// Block = 256 threads = 8 waves; each wave computes a 16x64 tile.
template <int KIN, bool TANH, bool MUL>
__global__ void __launch_bounds__(256)
linear_wmma(const float* __restrict__ in, const float* __restrict__ mul,
            const bf16* __restrict__ Wpk, const float* __restrict__ bias,
            float* __restrict__ out, int n) {
  const int tid     = threadIdx.x;
  const int lane    = tid & 31;
  const int wave    = tid >> 5;
  const int rowBase = blockIdx.x * 128 + wave * 16;
  if (rowBase >= n) return;             // wave-uniform: n % 16 == 0 here

  const int half = lane >> 4;           // 0: lanes 0-15, 1: lanes 16-31
  const int m    = lane & 15;
  const int row  = rowBase + m;

  v8f c0 = {}, c1 = {}, c2 = {}, c3 = {};

  const float* __restrict__ rowp = in + (size_t)row * KIN;
  const float* __restrict__ mulp = MUL ? (mul + (size_t)row * DHID) : nullptr;
  const v16bf* __restrict__ Wf   = (const v16bf*)Wpk;

#pragma unroll
  for (int s = 0; s < (KIN + 31) / 32; ++s) {
    const int kb = 32 * s + (half ? 8 : 0);

    // ---- A fragment (16x32 bf16): vector loads + cvt ----
    v16bf a;
    if constexpr (KIN == 4) {
      const float4 v = *(const float4*)rowp;            // k = 0..3 only
      const float vv[4] = {v.x, v.y, v.z, v.w};
#pragma unroll
      for (int e = 0; e < 16; ++e)
        a[e] = (half == 0 && e < 4) ? (bf16)vv[e < 4 ? e : 0] : (bf16)0.0f;
    } else {
      const float4 q0 = *(const float4*)(rowp + kb);
      const float4 q1 = *(const float4*)(rowp + kb + 4);
      const float4 q2 = *(const float4*)(rowp + kb + 16);
      const float4 q3 = *(const float4*)(rowp + kb + 20);
      float va[16] = {q0.x, q0.y, q0.z, q0.w, q1.x, q1.y, q1.z, q1.w,
                      q2.x, q2.y, q2.z, q2.w, q3.x, q3.y, q3.z, q3.w};
      float vm[16];
      if constexpr (MUL) {
        const float4 p0 = *(const float4*)(mulp + kb);
        const float4 p1 = *(const float4*)(mulp + kb + 4);
        const float4 p2 = *(const float4*)(mulp + kb + 16);
        const float4 p3 = *(const float4*)(mulp + kb + 20);
        const float tmp[16] = {p0.x, p0.y, p0.z, p0.w, p1.x, p1.y, p1.z, p1.w,
                               p2.x, p2.y, p2.z, p2.w, p3.x, p3.y, p3.z, p3.w};
#pragma unroll
        for (int e = 0; e < 16; ++e) vm[e] = tmp[e];
      }
#pragma unroll
      for (int e = 0; e < 16; ++e) {
        float v = va[e];
        if (TANH) v = fast_tanh(v);
        if (MUL)  v *= vm[e];
        a[e] = (bf16)v;
      }
    }

    // ---- B fragments: one 32B vector load per tile ----
    const v16bf b0v = Wf[(s * 4 + 0) * 32 + lane];
    const v16bf b1v = Wf[(s * 4 + 1) * 32 + lane];
    const v16bf b2v = Wf[(s * 4 + 2) * 32 + lane];
    const v16bf b3v = Wf[(s * 4 + 3) * 32 + lane];

    c0 = __builtin_amdgcn_wmma_f32_16x16x32_bf16(false, a, false, b0v, (short)0, c0, false, false);
    c1 = __builtin_amdgcn_wmma_f32_16x16x32_bf16(false, a, false, b1v, (short)0, c1, false, false);
    c2 = __builtin_amdgcn_wmma_f32_16x16x32_bf16(false, a, false, b2v, (short)0, c2, false, false);
    c3 = __builtin_amdgcn_wmma_f32_16x16x32_bf16(false, a, false, b3v, (short)0, c3, false, false);
  }

  const float bm0 = bias[m];
  const float bm1 = bias[m + 16];
  const float bm2 = bias[m + 32];
  const float bm3 = bias[m + 48];

  // C/D layout: VGPR v -> M = v + 8*half, N = lane%16 (+16 per tile)
#pragma unroll
  for (int v = 0; v < 8; ++v) {
    const int M = (half ? 8 : 0) + v;
    float* op = out + (size_t)(rowBase + M) * DHID + m;
    op[0]  = c0[v] + bm0;
    op[16] = c1[v] + bm1;
    op[32] = c2[v] + bm2;
    op[48] = c3[v] + bm3;
  }
}

// ---- normalized gather + scatter-add over edges (+ implicit self loops) ---
__global__ void scatter_edges(const float* __restrict__ hlin, const float* __restrict__ dis,
                              const int* __restrict__ row, const int* __restrict__ col,
                              float* __restrict__ acc, int E, int n) {
  size_t t = (size_t)blockIdx.x * blockDim.x + threadIdx.x;
  size_t total = (size_t)(E + n) * 16;          // 16 threads/edge, 4 floats each
  if (t >= total) return;
  int e = (int)(t >> 4);
  int f = (int)(t & 15) * 4;
  int r, c;
  if (e < E) { r = row[e]; c = col[e]; } else { r = c = e - E; }
  float nrm = dis[r] * dis[c];
  const float4 v = *(const float4*)(hlin + (size_t)r * DHID + f);
  float* dst = acc + (size_t)c * DHID + f;
  atomicAdd(dst + 0, nrm * v.x);
  atomicAdd(dst + 1, nrm * v.y);
  atomicAdd(dst + 2, nrm * v.z);
  atomicAdd(dst + 3, nrm * v.w);
}

// ---- graph pooling ---------------------------------------------------------
__global__ void pool_accum(const float* __restrict__ h, const int* __restrict__ batch,
                           unsigned* __restrict__ pmax, float* __restrict__ psum,
                           float* __restrict__ cnt, int n) {
  size_t t = (size_t)blockIdx.x * blockDim.x + threadIdx.x;
  if (t >= (size_t)n * DHID) return;
  int node = (int)(t >> 6);
  int f    = (int)(t & 63);
  int g    = batch[node];
  float v  = h[t];
  atomicMax(&pmax[g * DHID + f], f2ord(v));
  atomicAdd(&psum[g * DHID + f], v);
  if (f == 0) atomicAdd(&cnt[g], 1.0f);
}

__global__ void pool_finalize(const unsigned* __restrict__ pmax, const float* __restrict__ psum,
                              const float* __restrict__ cnt, float* __restrict__ pooled, int G) {
  int t = blockIdx.x * blockDim.x + threadIdx.x;
  if (t >= G * 2 * DHID) return;
  int g = t / (2 * DHID);
  int f = t % (2 * DHID);
  pooled[t] = (f < DHID) ? ord2f(pmax[g * DHID + f])
                         : psum[g * DHID + (f - DHID)] / cnt[g];
}

__global__ void out_proj(const float* __restrict__ pooled, const float* __restrict__ Wo,
                         const float* __restrict__ bo, float* __restrict__ outp,
                         int G, int FOUT) {
  int t = blockIdx.x * blockDim.x + threadIdx.x;
  if (t >= G * FOUT) return;
  int g = t / FOUT, j = t % FOUT;
  float s = bo[j];
  const float* pp = pooled + (size_t)g * 2 * DHID;
#pragma unroll 8
  for (int k = 0; k < 2 * DHID; ++k) s += pp[k] * Wo[k * FOUT + j];
  outp[t] = s;
}

// ---------------------------------------------------------------------------
extern "C" void kernel_launch(void* const* d_in, const int* in_sizes, int n_in,
                              void* d_out, int out_size, void* d_ws, size_t ws_size,
                              hipStream_t stream) {
  (void)n_in; (void)ws_size;
  const float* x    = (const float*)d_in[0];
  const int*   ei   = (const int*)d_in[1];
  const int*   bi   = (const int*)d_in[2];
  const float* topo = (const float*)d_in[3];
  const float* W0 = (const float*)d_in[4];   const float* b0 = (const float*)d_in[5];
  const float* W1 = (const float*)d_in[6];   const float* b1 = (const float*)d_in[7];
  const float* W2 = (const float*)d_in[8];   const float* b2 = (const float*)d_in[9];
  const float* W3 = (const float*)d_in[10];  const float* b3 = (const float*)d_in[11];
  const float* Wfu = (const float*)d_in[12]; const float* bfu = (const float*)d_in[13];
  const float* Wo = (const float*)d_in[14];  const float* bo = (const float*)d_in[15];

  const int N    = in_sizes[0] / 4;            // F_IN = 4
  const int E    = in_sizes[1] / 2;            // edge_index is [2, E]
  const int FOUT = in_sizes[15];               // bo
  const int G    = out_size / (FOUT + 2 * DHID);

  const int* row = ei;                         // sources
  const int* col = ei + E;                     // targets

  float* ws = (float*)d_ws;
  size_t off = 0;
  float*    dis  = ws + off; off += ((size_t)N + 63) & ~(size_t)63;
  float*    hA   = ws + off; off += (size_t)N * DHID;
  float*    hlin = ws + off; off += (size_t)N * DHID;
  float*    hB   = ws + off; off += (size_t)N * DHID;
  unsigned* pmax = (unsigned*)(ws + off); off += (size_t)G * DHID;
  float*    psum = ws + off; off += (size_t)G * DHID;
  float*    cnt  = ws + off; off += ((size_t)G + 63) & ~(size_t)63;
  bf16*     p0   = (bf16*)(ws + off); off += 1024;   // 4 frags * 512 bf16
  bf16*     p1   = (bf16*)(ws + off); off += 2048;   // 8 frags * 512 bf16
  bf16*     p2   = (bf16*)(ws + off); off += 2048;
  bf16*     p3   = (bf16*)(ws + off); off += 2048;
  bf16*     p4   = (bf16*)(ws + off); off += 2048;

  auto cdiv = [](size_t a, size_t b) { return (unsigned)((a + b - 1) / b); };

  // --- weight pre-swizzle into fragment order ---
  pack_weights<4> <<<cdiv(2048, 256), 256, 0, stream>>>(W0,  p0);
  pack_weights<64><<<cdiv(4096, 256), 256, 0, stream>>>(W1,  p1);
  pack_weights<64><<<cdiv(4096, 256), 256, 0, stream>>>(W2,  p2);
  pack_weights<64><<<cdiv(4096, 256), 256, 0, stream>>>(W3,  p3);
  pack_weights<64><<<cdiv(4096, 256), 256, 0, stream>>>(Wfu, p4);

  // --- degree / normalization ---
  deg_init <<<cdiv(N, 256), 256, 0, stream>>>(dis, N);
  deg_accum<<<cdiv(E, 256), 256, 0, stream>>>(col, dis, E);
  deg_rsqrt<<<cdiv(N, 256), 256, 0, stream>>>(dis, N);

  const unsigned lin_grid = cdiv(N, 128);
  const size_t   ND = (size_t)N * DHID;
  const size_t   scat = (size_t)(E + N) * 16;

  // --- layer 0 : x @ W0 + b0, scatter ---
  linear_wmma<4,  false, false><<<lin_grid, 256, 0, stream>>>(x,  nullptr, p0, b0, hlin, N);
  fill_u32<<<cdiv(ND, 256), 256, 0, stream>>>((unsigned*)hA, 0u, ND);
  scatter_edges<<<cdiv(scat, 256), 256, 0, stream>>>(hlin, dis, row, col, hA, E, N);

  // --- layer 1 : tanh fused into A-load ---
  linear_wmma<64, true,  false><<<lin_grid, 256, 0, stream>>>(hA, nullptr, p1, b1, hlin, N);
  fill_u32<<<cdiv(ND, 256), 256, 0, stream>>>((unsigned*)hB, 0u, ND);
  scatter_edges<<<cdiv(scat, 256), 256, 0, stream>>>(hlin, dis, row, col, hB, E, N);

  // --- layer 2 ---
  linear_wmma<64, true,  false><<<lin_grid, 256, 0, stream>>>(hB, nullptr, p2, b2, hlin, N);
  fill_u32<<<cdiv(ND, 256), 256, 0, stream>>>((unsigned*)hA, 0u, ND);
  scatter_edges<<<cdiv(scat, 256), 256, 0, stream>>>(hlin, dis, row, col, hA, E, N);

  // --- layer 3 ---
  linear_wmma<64, true,  false><<<lin_grid, 256, 0, stream>>>(hA, nullptr, p3, b3, hlin, N);
  fill_u32<<<cdiv(ND, 256), 256, 0, stream>>>((unsigned*)hB, 0u, ND);
  scatter_edges<<<cdiv(scat, 256), 256, 0, stream>>>(hlin, dis, row, col, hB, E, N);

  // --- fusion : (topo * tanh(h)) @ Wf + bf ---
  linear_wmma<64, true,  true ><<<lin_grid, 256, 0, stream>>>(hB, topo, p4, bfu, hlin, N);

  // --- pooling (max + mean per graph) ---
  fill_u32<<<cdiv((size_t)G * (2 * DHID + 1), 256), 256, 0, stream>>>(pmax, 0u,
                                                 (size_t)G * (2 * DHID + 1));
  pool_accum<<<cdiv(ND, 256), 256, 0, stream>>>(hlin, bi, pmax, psum, cnt, N);

  float* outp   = (float*)d_out;               // [G, FOUT] first
  float* pooled = outp + (size_t)G * FOUT;     // then [G, 2*DHID]
  pool_finalize<<<cdiv((size_t)G * 2 * DHID, 256), 256, 0, stream>>>(pmax, psum, cnt, pooled, G);
  out_proj<<<cdiv((size_t)G * FOUT, 256), 256, 0, stream>>>(pooled, Wo, bo, outp, G, FOUT);
}
